// AutoEncoder_14328010899797
// MI455X (gfx1250) — compile-verified, matrix operand order checked by reading, not run on hardware
//
#include <hip/hip_runtime.h>
#include <cstdint>

// ---------------------------------------------------------------------------
// MI455X (gfx1250) fused set-autoencoder.
// ~59 GFLOP of skinny GEMMs vs ~260MB HBM -> compute bound; all Linears run on
// v_wmma_f32_16x16x32_f16 (f16 in, f32 acc). Segment softmax/sum via f32
// global atomics (segment tables ~4MB, L2-resident). Weights pre-swizzled in
// global memory into per-lane WMMA B-fragment order: one fragment = 32 lanes
// x 8 contiguous dwords -> each lane loads its fragment with 2x b128.
// ---------------------------------------------------------------------------

#define NELEM 500000
#define BSEG  5000
#define NTILES (NELEM / 16)   // 31250, N is a multiple of 16

typedef __attribute__((ext_vector_type(16))) _Float16 v16h;
typedef __attribute__((ext_vector_type(8)))  float    v8f;

// ---- workspace layout (units of 4 bytes) ----------------------------------
enum : int {
  OFF_SEGMAX = 0,            // B*4 f32, init -inf
  OFF_SEGSUM = 20000,        // B*4 f32, init 0
  OFF_KEYBUF = 40000,        // N*4 f32 (key0, then exp values)
  OFF_Z      = 2040000,      // B*128 f32, init 0
  OFF_KEYENC = 2680000,      // B*64 f32, init 0
  OFF_W      = 3000000,      // packed weights (u32 dwords) below
  OW_EKW1 = 0,      OW_EKW2 = 1280,  OW_EVW1 = 7424,  OW_EVW2 = 10496,
  OW_KSEW1 = 16640, OW_KSEW2 = 17408, OW_KSDW1 = 19456, OW_KSDW2 = 20992,
  OW_DKW1 = 21504,  OW_DKW2 = 22784, OW_DDW1 = 28928,  OW_DDW2 = 35072,
  WS_INIT_DWORDS = 3000000
};

// ---- small helpers --------------------------------------------------------
__device__ __forceinline__ uint16_t f2h(float x) {
  union { _Float16 h; uint16_t u; } c; c.h = (_Float16)x; return c.u;
}
__device__ __forceinline__ float mishf(float x) {
  float sp = (x > 20.f) ? x : log1pf(__expf(x));   // softplus, stable
  return x * tanhf(sp);
}
__device__ __forceinline__ void atomicMaxF(float* a, float v) {
  // IEEE monotonic-bits trick (segmax initialized to -inf = 0xFF800000)
  if (v >= 0.f) atomicMax((int*)a, __float_as_int(v));
  else          atomicMin((unsigned int*)a, __float_as_uint(v));
}

// ---- WMMA fragment loaders (16x16x32 f16, wave32 layouts per ISA 7.12.2) --
// A (16x32 MxK): lane m=lane&15, h=lane>>4; VGPR p holds K pair
// k0(p) = (p>>2)*16 + h*8 + (p&3)*2  -> p0..3 / p4..7 are 16B-contiguous in K.
__device__ __forceinline__ v16h load_a(const uint16_t* a, int aS, int kOff, int lane) {
  int m = lane & 15, h = lane >> 4;
  union { v16h v; uint4 q[2]; } f;
  const uint16_t* p = a + m * aS + kOff + h * 8;
  f.q[0] = *(const uint4*)(p);        // ds_load_b128
  f.q[1] = *(const uint4*)(p + 16);   // ds_load_b128
  return f.v;
}
// B fragments pre-swizzled per lane: frag index (kt*NT + t), 32 lanes x 8
// contiguous dwords -> two global_load_b128 per fragment, fully coalesced.
__device__ __forceinline__ v16h load_b_frag(const uint32_t* __restrict__ w,
                                            int frag, int lane) {
  const uint32_t* p = w + (((size_t)frag << 5) + lane) * 8;
  union { v16h v; uint4 q[2]; } f;
  f.q[0] = *(const uint4*)(p);
  f.q[1] = *(const uint4*)(p + 4);
  return f.v;
}

// ---- GEMM tiles: D[16xNPAD] = A_lds[16xKPAD] x Wpacked[KPADxNPAD] ---------
template<int KPAD, int NPAD>
__device__ __forceinline__ void gemm_acc(const uint16_t* aLDS, int aS,
                                         const uint32_t* __restrict__ wG,
                                         v8f* acc, int lane) {
  constexpr int KT = KPAD / 32, NT = NPAD / 16;
  const v8f vzero = {0.f, 0.f, 0.f, 0.f, 0.f, 0.f, 0.f, 0.f};
#pragma unroll
  for (int t = 0; t < NT; ++t) acc[t] = vzero;
#pragma unroll
  for (int kk = 0; kk < KT; ++kk) {
    v16h a = load_a(aLDS, aS, kk * 32, lane);
#pragma unroll
    for (int t = 0; t < NT; ++t) {
      v16h b = load_b_frag(wG, kk * NT + t, lane);
      acc[t] = __builtin_amdgcn_wmma_f32_16x16x32_f16(
          false, a, false, b, (short)0, acc[t], false, false);
    }
  }
}

template<int KPAD, int NPAD>
__device__ __forceinline__ void gemm_to_lds(const uint16_t* aLDS, int aS,
                                            const uint32_t* __restrict__ wG,
                                            float* cLDS, int cS, int lane) {
  constexpr int NT = NPAD / 16;
  v8f acc[NT];
  gemm_acc<KPAD, NPAD>(aLDS, aS, wG, acc, lane);
  int n = lane & 15, h = lane >> 4;   // C layout: lane=col, vgpr r -> row r+h*8
#pragma unroll
  for (int t = 0; t < NT; ++t)
#pragma unroll
    for (int r = 0; r < 8; ++r)
      cLDS[(r + h * 8) * cS + t * 16 + n] = acc[t][r];
}

// ---- staging: 16x64 f32 global rows -> f16 LDS tile (float4 + b64 stores) -
__device__ __forceinline__ void stage16x64(const float* __restrict__ src,
                                           const int* rowIdx,  // or nullptr
                                           int rowBase, uint16_t* dst, int lane) {
  for (int i = lane; i < 256; i += 32) {
    int r = i >> 4, c = (i & 15) << 2;
    const float* s = src + (size_t)(rowIdx ? rowIdx[r] : rowBase + r) * 64 + c;
    float4 v = *(const float4*)s;
    union { uint16_t h[4]; uint2 u; } pk;
    pk.h[0] = f2h(v.x); pk.h[1] = f2h(v.y); pk.h[2] = f2h(v.z); pk.h[3] = f2h(v.w);
    *(uint2*)(dst + r * 64 + c) = pk.u;
  }
}

// ---- epilogues: bias -> (LayerNorm) -> mish -> f16 A-tile, zero-padded ----
__device__ __forceinline__ void ep_ln_mish(const float* c, int cS,
    const float* __restrict__ b1, const float* __restrict__ g,
    const float* __restrict__ be, uint16_t* out, int oS,
    int Nreal, int Npad, int lane) {
  int row = lane & 15;                 // both half-waves compute redundantly
  float mean = 0.f;
  for (int j = 0; j < Nreal; ++j) mean += c[row * cS + j] + b1[j];
  mean /= (float)Nreal;
  float var = 0.f;
  for (int j = 0; j < Nreal; ++j) {
    float v = c[row * cS + j] + b1[j] - mean; var += v * v;
  }
  var /= (float)Nreal;                 // jnp.var: biased
  float rstd = rsqrtf(var + 1e-5f);
  if (lane < 16) {
    for (int j = 0; j < Nreal; ++j) {
      float v = (c[row * cS + j] + b1[j] - mean) * rstd * g[j] + be[j];
      out[row * oS + j] = f2h(mishf(v));
    }
    for (int j = Nreal; j < Npad; ++j) out[row * oS + j] = 0;
  }
}
__device__ __forceinline__ void ep_mish(const float* c, int cS,
    const float* __restrict__ b1, uint16_t* out, int oS,
    int Nreal, int Npad, int lane) {
  if (lane < 16) {
    int row = lane;
    for (int j = 0; j < Nreal; ++j)
      out[row * oS + j] = f2h(mishf(c[row * cS + j] + b1[j]));
    for (int j = Nreal; j < Npad; ++j) out[row * oS + j] = 0;
  }
}

// ---------------------------------------------------------------------------
// K0: init segment scratch (deterministic per launch)
__global__ void k_init(float* ws) {
  int i = blockIdx.x * blockDim.x + threadIdx.x;
  if (i >= WS_INIT_DWORDS) return;
  ((uint32_t*)ws)[i] = (i < OFF_SEGSUM) ? 0xFF800000u /* -inf */ : 0u;
}

// K-pack: f32 row-major [Kreal x Nreal] -> per-lane-swizzled f16 B fragments.
// dst[i]: p = i&7 (dword in lane frag), lane = (i>>3)&31, frag = i>>8,
// frag = kt*NT + t; dword = {W[k0][n], W[k0+1][n]} with the WMMA B mapping.
__global__ void k_pack(const float* __restrict__ W, int Kreal, int Nreal,
                       uint32_t* __restrict__ dst, int KPAD, int NPAD) {
  int i = blockIdx.x * blockDim.x + threadIdx.x;
  int total = (KPAD / 2) * NPAD;
  if (i >= total) return;
  int p = i & 7, lane = (i >> 3) & 31, frag = i >> 8;
  int NT = NPAD / 16;
  int kt = frag / NT, t = frag % NT;
  int n = (lane & 15) + t * 16, h = lane >> 4;
  int k0 = kt * 32 + ((p >> 2) << 4) + (h << 3) + ((p & 3) << 1);
  _Float16 lo = (k0 < Kreal && n < Nreal) ? (_Float16)W[k0 * Nreal + n] : (_Float16)0.f;
  _Float16 hi = (k0 + 1 < Kreal && n < Nreal) ? (_Float16)W[(k0 + 1) * Nreal + n] : (_Float16)0.f;
  union { _Float16 h[2]; uint32_t u; } pk; pk.h[0] = lo; pk.h[1] = hi;
  dst[i] = pk.u;
}

// K1: key0 = x @ W_map + b_map ; atomic segment max
__global__ void k_key0(const float* __restrict__ x, const int* __restrict__ batch,
                       const float* __restrict__ Wm, const float* __restrict__ bm,
                       float* keybuf, float* segmax) {
  int i = blockIdx.x * blockDim.x + threadIdx.x;
  if (i >= NELEM) return;
  float a0 = bm[0], a1 = bm[1], a2 = bm[2], a3 = bm[3];
  const float* xr = x + (size_t)i * 64;
#pragma unroll 8
  for (int k = 0; k < 64; ++k) {
    float xv = xr[k];
    a0 += xv * Wm[k * 4 + 0]; a1 += xv * Wm[k * 4 + 1];
    a2 += xv * Wm[k * 4 + 2]; a3 += xv * Wm[k * 4 + 3];
  }
  int b = batch[i];
  keybuf[i * 4 + 0] = a0; keybuf[i * 4 + 1] = a1;
  keybuf[i * 4 + 2] = a2; keybuf[i * 4 + 3] = a3;
  atomicMaxF(&segmax[b * 4 + 0], a0); atomicMaxF(&segmax[b * 4 + 1], a1);
  atomicMaxF(&segmax[b * 4 + 2], a2); atomicMaxF(&segmax[b * 4 + 3], a3);
}

// K2: e = exp(key0 - segmax) ; atomic segment sum
__global__ void k_expsum(const float* __restrict__ segmax, const int* __restrict__ batch,
                         float* keybuf, float* segsum) {
  int i = blockIdx.x * blockDim.x + threadIdx.x;
  if (i >= NELEM) return;
  int b = batch[i];
#pragma unroll
  for (int f = 0; f < 4; ++f) {
    float e = __expf(keybuf[i * 4 + f] - segmax[b * 4 + f]);
    keybuf[i * 4 + f] = e;
    atomicAdd(&segsum[b * 4 + f], e);
  }
}

// ---------------------------------------------------------------------------
struct EncArgs {
  const float* x; const int* batch;
  const float* keybuf; const float* segsum;
  float* z; float* keyenc;
  const uint32_t *ekW1, *ekW2, *evW1, *evW2, *kseW1, *kseW2;
  const float *ek_b1, *ek_g, *ek_be, *ek_b2;
  const float *ev_b1, *ev_g, *ev_be, *ev_b2;
  const float *kse_b1, *kse_g, *kse_be, *kse_b2;
};

__global__ __launch_bounds__(128) void k_encoder(EncArgs A) {
  __shared__ alignas(16) uint16_t s_kt[4][16 * 32];
  __shared__ alignas(16) uint16_t s_ab[4][16 * 128];
  __shared__ float s_cb[4][16 * 128];
  __shared__ int s_bid[4][16];
  int wave = threadIdx.x >> 5, lane = threadIdx.x & 31;
  uint16_t* kt = s_kt[wave]; uint16_t* ab = s_ab[wave];
  float* cb = s_cb[wave];    int* bid = s_bid[wave];
  int tile = blockIdx.x * 4 + wave;
  bool active = tile < NTILES;              // wave-uniform; inactive waves
  int rowBase = active ? tile * 16 : 0;     // compute tile 0, suppress stores

  __builtin_prefetch(A.x + (size_t)(rowBase + 16) * 64, 0, 1);

  if (lane < 16) bid[lane] = A.batch[rowBase + lane];
  for (int i = lane; i < 16 * 32; i += 32) kt[i] = 0;
  __syncthreads();
  for (int i = lane; i < 64; i += 32) {     // keys = e / segsum, f16, pad K->32
    int r = i >> 2, f = i & 3, b = bid[r];
    kt[r * 32 + f] = f2h(A.keybuf[(rowBase + r) * 4 + f] / A.segsum[b * 4 + f]);
  }
  __syncthreads();

  // ---- gate = mlp2_ln(keys): [16x4->66] LN mish -> [16x128] --------------
  gemm_to_lds<32, 80>(kt, 32, A.ekW1, cb, 80, lane);
  __syncthreads();
  ep_ln_mish(cb, 80, A.ek_b1, A.ek_g, A.ek_be, ab, 96, 66, 96, lane);
  __syncthreads();
  v8f gate[8];
  gemm_acc<96, 128>(ab, 96, A.ekW2, gate, lane);
  {
    int n = lane & 15;
#pragma unroll
    for (int t = 0; t < 8; ++t) {
      float bb = A.ek_b2[t * 16 + n];
#pragma unroll
      for (int r = 0; r < 8; ++r) gate[t][r] += bb;
    }
  }
  __syncthreads();

  // ---- vals = mlp2_ln(x): [16x64->96] LN mish -> [16x128] ----------------
  stage16x64(A.x, nullptr, rowBase, ab, lane);
  __syncthreads();
  gemm_to_lds<64, 96>(ab, 64, A.evW1, cb, 96, lane);
  __syncthreads();
  ep_ln_mish(cb, 96, A.ev_b1, A.ev_g, A.ev_be, ab, 96, 96, 96, lane);
  __syncthreads();
  v8f val[8];
  gemm_acc<96, 128>(ab, 96, A.evW2, val, lane);

  if (active) {                              // y = vals*gate -> z[batch]
    int n = lane & 15, h = lane >> 4;
#pragma unroll
    for (int t = 0; t < 8; ++t) {
      float bb = A.ev_b2[t * 16 + n];
#pragma unroll
      for (int r = 0; r < 8; ++r) {
        float y = (val[t][r] + bb) * gate[t][r];
        atomicAdd(&A.z[bid[r + h * 8] * 128 + t * 16 + n], y);
      }
    }
  }
  __syncthreads();

  // ---- keyenc += mlp2_ln(keys) via key_storage encoder -------------------
  gemm_to_lds<32, 48>(kt, 32, A.kseW1, cb, 48, lane);
  __syncthreads();
  ep_ln_mish(cb, 48, A.kse_b1, A.kse_g, A.kse_be, ab, 64, 34, 64, lane);
  __syncthreads();
  v8f ke[4];
  gemm_acc<64, 64>(ab, 64, A.kseW2, ke, lane);
  if (active) {
    int n = lane & 15, h = lane >> 4;
#pragma unroll
    for (int t = 0; t < 4; ++t) {
      float bb = A.kse_b2[t * 16 + n];
#pragma unroll
      for (int r = 0; r < 8; ++r)
        atomicAdd(&A.keyenc[bid[r + h * 8] * 64 + t * 16 + n], ke[t][r] + bb);
    }
  }
}

// ---------------------------------------------------------------------------
struct DecArgs {
  const int* batch;
  const float* z; const float* keyenc;
  float* out;
  const uint32_t *ksdW1, *ksdW2, *dkW1, *dkW2, *ddW1, *ddW2;
  const float *ksd_b1, *ksd_g, *ksd_be, *ksd_b2;
  const float *dk_b1, *dk_g, *dk_be, *dk_b2;
  const float *dd_b1, *dd_b2;
};

__global__ __launch_bounds__(128) void k_decoder(DecArgs A) {
  __shared__ alignas(16) uint16_t s_kt[4][16 * 32];
  __shared__ alignas(16) uint16_t s_ab[4][16 * 128];
  __shared__ float s_cb[4][16 * 128];
  __shared__ int s_bid[4][16];
  int wave = threadIdx.x >> 5, lane = threadIdx.x & 31;
  uint16_t* kt = s_kt[wave]; uint16_t* ab = s_ab[wave];
  float* cb = s_cb[wave];    int* bid = s_bid[wave];
  int tile = blockIdx.x * 4 + wave;
  bool active = tile < NTILES;
  int rowBase = active ? tile * 16 : 0;

  if (lane < 16) bid[lane] = A.batch[rowBase + lane];
  __syncthreads();

  // ---- keyenc gather (= repeat) -> A tile [16x64] -------------------------
  stage16x64(A.keyenc, bid, 0, ab, lane);
  __syncthreads();
  // ---- keys_rec = mlp2_ln(keyenc): 64->34(LN mish)->4 ---------------------
  gemm_to_lds<64, 48>(ab, 64, A.ksdW1, cb, 48, lane);
  __syncthreads();
  ep_ln_mish(cb, 48, A.ksd_b1, A.ksd_g, A.ksd_be, ab, 64, 34, 64, lane);
  __syncthreads();
  v8f kr[1];
  gemm_acc<64, 16>(ab, 64, A.ksdW2, kr, lane);
  for (int i = lane; i < 16 * 32; i += 32) kt[i] = 0;
  __syncthreads();
  {
    int n = lane & 15, h = lane >> 4;
    if (n < 4) {
      float bb = A.ksd_b2[n];
#pragma unroll
      for (int r = 0; r < 8; ++r)
        kt[(r + h * 8) * 32 + n] = f2h(kr[0][r] + bb);
    }
  }
  __syncthreads();

  // ---- q = mlp2_ln(keys_rec): 4->66(LN mish)->128 -------------------------
  gemm_to_lds<32, 80>(kt, 32, A.dkW1, cb, 80, lane);
  __syncthreads();
  ep_ln_mish(cb, 80, A.dk_b1, A.dk_g, A.dk_be, ab, 96, 66, 96, lane);
  __syncthreads();
  v8f q[8];
  gemm_acc<96, 128>(ab, 96, A.dkW2, q, lane);
  __syncthreads();

  // ---- zp = z[batch] * q -> A tile [16x128] -------------------------------
  {
    int n = lane & 15, h = lane >> 4;
#pragma unroll
    for (int t = 0; t < 8; ++t) {
      float bb = A.dk_b2[t * 16 + n];
#pragma unroll
      for (int r = 0; r < 8; ++r) {
        float zz = A.z[bid[r + h * 8] * 128 + t * 16 + n];
        ab[(r + h * 8) * 128 + t * 16 + n] = f2h(zz * (q[t][r] + bb));
      }
    }
  }
  __syncthreads();

  // ---- xr = mlp2(zp): 128->96(mish)->64 -----------------------------------
  gemm_to_lds<128, 96>(ab, 128, A.ddW1, cb, 96, lane);
  __syncthreads();
  ep_mish(cb, 96, A.dd_b1, ab, 96, 96, 96, lane);
  __syncthreads();
  v8f o[4];
  gemm_acc<96, 64>(ab, 96, A.ddW2, o, lane);
  if (active) {
    int n = lane & 15, h = lane >> 4;
#pragma unroll
    for (int t = 0; t < 4; ++t) {
      float bb = A.dd_b2[t * 16 + n];
#pragma unroll
      for (int r = 0; r < 8; ++r)
        A.out[(size_t)(rowBase + r + h * 8) * 64 + t * 16 + n] = o[t][r] + bb;
    }
    if (lane < 16)  // batch_rec == batch (batch is sorted)
      ((int*)A.out)[(size_t)NELEM * 64 + rowBase + lane] = bid[lane];
  }
}

// ---------------------------------------------------------------------------
extern "C" void kernel_launch(void* const* d_in, const int* in_sizes, int n_in,
                              void* d_out, int out_size, void* d_ws, size_t ws_size,
                              hipStream_t stream) {
  (void)in_sizes; (void)n_in; (void)out_size; (void)ws_size;
  const float* x     = (const float*)d_in[0];
  const int*   batch = (const int*)d_in[1];
  // d_in[2] = n (unused: batch is sorted, gather by batch == repeat by n)
  float* ws = (float*)d_ws;
  float* segmax = ws + OFF_SEGMAX;
  float* segsum = ws + OFF_SEGSUM;
  float* keybuf = ws + OFF_KEYBUF;
  float* z      = ws + OFF_Z;
  float* keyenc = ws + OFF_KEYENC;
  uint32_t* wp  = (uint32_t*)(ws + OFF_W);

  k_init<<<(WS_INIT_DWORDS + 255) / 256, 256, 0, stream>>>(ws);

  struct { const float* W; int K, N, KP, NP; uint32_t* dst; } jobs[12] = {
    { (const float*)d_in[5],   4,  66, 32,  80, wp + OW_EKW1 },
    { (const float*)d_in[9],  66, 128, 96, 128, wp + OW_EKW2 },
    { (const float*)d_in[11], 64,  96, 64,  96, wp + OW_EVW1 },
    { (const float*)d_in[15], 96, 128, 96, 128, wp + OW_EVW2 },
    { (const float*)d_in[17],  4,  34, 32,  48, wp + OW_KSEW1 },
    { (const float*)d_in[21], 34,  64, 64,  64, wp + OW_KSEW2 },
    { (const float*)d_in[23], 64,  34, 64,  48, wp + OW_KSDW1 },
    { (const float*)d_in[27], 34,   4, 64,  16, wp + OW_KSDW2 },
    { (const float*)d_in[29],  4,  66, 32,  80, wp + OW_DKW1 },
    { (const float*)d_in[33], 66, 128, 96, 128, wp + OW_DKW2 },
    { (const float*)d_in[35],128,  96,128,  96, wp + OW_DDW1 },
    { (const float*)d_in[37], 96,  64, 96,  64, wp + OW_DDW2 },
  };
  for (auto& j : jobs) {
    int total = (j.KP / 2) * j.NP;
    k_pack<<<(total + 255) / 256, 256, 0, stream>>>(j.W, j.K, j.N, j.dst, j.KP, j.NP);
  }

  k_key0<<<(NELEM + 255) / 256, 256, 0, stream>>>(
      x, batch, (const float*)d_in[3], (const float*)d_in[4], keybuf, segmax);
  k_expsum<<<(NELEM + 255) / 256, 256, 0, stream>>>(segmax, batch, keybuf, segsum);

  EncArgs ea;
  ea.x = x; ea.batch = batch; ea.keybuf = keybuf; ea.segsum = segsum;
  ea.z = z; ea.keyenc = keyenc;
  ea.ekW1 = wp + OW_EKW1; ea.ekW2 = wp + OW_EKW2;
  ea.evW1 = wp + OW_EVW1; ea.evW2 = wp + OW_EVW2;
  ea.kseW1 = wp + OW_KSEW1; ea.kseW2 = wp + OW_KSEW2;
  ea.ek_b1 = (const float*)d_in[6];  ea.ek_g = (const float*)d_in[7];
  ea.ek_be = (const float*)d_in[8];  ea.ek_b2 = (const float*)d_in[10];
  ea.ev_b1 = (const float*)d_in[12]; ea.ev_g = (const float*)d_in[13];
  ea.ev_be = (const float*)d_in[14]; ea.ev_b2 = (const float*)d_in[16];
  ea.kse_b1 = (const float*)d_in[18]; ea.kse_g = (const float*)d_in[19];
  ea.kse_be = (const float*)d_in[20]; ea.kse_b2 = (const float*)d_in[22];

  int blocks = (NTILES + 3) / 4;   // 4 waves/block, 1 tile of 16 rows per wave
  k_encoder<<<blocks, 128, 0, stream>>>(ea);

  DecArgs da;
  da.batch = batch; da.z = z; da.keyenc = keyenc; da.out = (float*)d_out;
  da.ksdW1 = wp + OW_KSDW1; da.ksdW2 = wp + OW_KSDW2;
  da.dkW1 = wp + OW_DKW1;   da.dkW2 = wp + OW_DKW2;
  da.ddW1 = wp + OW_DDW1;   da.ddW2 = wp + OW_DDW2;
  da.ksd_b1 = (const float*)d_in[24]; da.ksd_g = (const float*)d_in[25];
  da.ksd_be = (const float*)d_in[26]; da.ksd_b2 = (const float*)d_in[28];
  da.dk_b1 = (const float*)d_in[30];  da.dk_g = (const float*)d_in[31];
  da.dk_be = (const float*)d_in[32];  da.dk_b2 = (const float*)d_in[34];
  da.dd_b1 = (const float*)d_in[36];  da.dd_b2 = (const float*)d_in[38];
  k_decoder<<<blocks, 128, 0, stream>>>(da);
}
